// LSTMModel_30855045054818
// MI455X (gfx1250) — matrix operand order repaired and checked by reading, not compiled, not run on hardware
//
#include <hip/hip_runtime.h>
#include <hip/hip_bf16.h>
#include <stdint.h>

// ---------------------------------------------------------------------------
// 2-layer LSTM (B=256, T=2048, I=8, H=100) fused persistent kernel for
// MI455X / gfx1250.  16 workgroups x 256 threads; each WG owns 16 batch rows
// for the entire sequence.  All weights LDS-resident (CDNA5 320KB WGP LDS).
// Matmuls: v_wmma_f32_16x16x32_bf16 (K=96 via 3 chunks) + f32 VALU tail for
// K=96..99.  Weight staging: TDM tensor_load_to_lds (TENSORcnt) for the
// fragment region + global_load_async_to_lds_b128 (ASYNCcnt) for the tail.
// ---------------------------------------------------------------------------

typedef __attribute__((ext_vector_type(16))) __bf16 v16bf;
typedef __attribute__((ext_vector_type(8)))  float  v8f;
typedef __attribute__((ext_vector_type(4)))  float  v4f;
typedef __attribute__((ext_vector_type(4)))  unsigned int v4u;
typedef __attribute__((ext_vector_type(8)))  int v8i;
typedef __attribute__((ext_vector_type(4)))  int v4i;

#define HDIM   100
#define TLEN   2048
#define GATE   400
#define K96    96

// LDS layout (byte offsets from dynamic-shared base; no static __shared__)
#define LDS_WFRAG 0u        // 3 x 76800 B  bf16 B-fragments (whh1, wih2, whh2)
#define LDS_WTAIL 230400u   // 3 x 4 x 400 f32  K-tail rows (19200 B)
#define LDS_WIH1  249600u   // 400 x 8 f32 (12800 B)
#define LDS_B1    262400u   // 400 f32 combined bias layer1
#define LDS_B2    264000u   // 400 f32 combined bias layer2
#define LDS_H1F   265600u   // 16 x 100 f32 h1
#define LDS_C1    272000u   // 16 x 100 f32 c1
#define LDS_H2F   278400u   // 16 x 100 f32 h2
#define LDS_C2    284800u   // 16 x 100 f32 c2
#define LDS_H1B   291200u   // 3 chunks x 32 lanes x 32B bf16 h1 A-fragments (3072 B)
#define LDS_H2B   294272u   // same for h2 (3072 B)
#define LDS_XROW  297344u   // 16 x 8 f32 current x tile (512 B)
#define LDS_G     297856u   // 16 x 400 f32 gate staging (25600 B)
#define LDS_TOTAL 323456u
#define WS_WBYTES 249600u   // bytes staged ws -> LDS (WFRAG + WTAIL)
#define WFRAG_BYTES 230400u

#if __has_builtin(__builtin_amdgcn_tensor_load_to_lds)
#define HAVE_TDM 1
#else
#define HAVE_TDM 0
#endif

__device__ __forceinline__ float tanh_fast(float x) {
#if __has_builtin(__builtin_amdgcn_tanhf)
  return __builtin_amdgcn_tanhf(x);        // v_tanh_f32 (gfx1250)
#else
  return 2.f / (1.f + __expf(-2.f * x)) - 1.f;
#endif
}
__device__ __forceinline__ float sigm(float x) {
  return 0.5f * tanh_fast(0.5f * x) + 0.5f;
}

// ---------------------------------------------------------------------------
// Prep: scatter f32 weights into bf16 WMMA-B-fragment order + f32 tail rows.
// ws block layout mirrors the LDS [WFRAG | WTAIL] region exactly.
// B-fragment addressing (bf16 units): w*38400 + ((c*25 + tile)*32 + lane)*16 + j
//   where lane = (k%32/16)*16 + (n%16), j = k%16, c = k/32, tile = n/16.
// ---------------------------------------------------------------------------
extern "C" __global__ void __launch_bounds__(256)
lstm_prep(const float* __restrict__ whh1, const float* __restrict__ wih2,
          const float* __restrict__ whh2, char* __restrict__ ws)
{
  int idx = blockIdx.x * 256 + threadIdx.x;
  if (idx < 3 * K96 * GATE) {
    int w = idx / (K96 * GATE);
    int r = idx % (K96 * GATE);
    int k = r / GATE;
    int n = r % GATE;
    const float* W = (w == 0) ? whh1 : (w == 1) ? wih2 : whh2;
    float v = W[n * HDIM + k];                    // B[k][n] = W[n][k]
    int c     = k >> 5;
    int kk    = k & 31;
    int khalf = kk >> 4;
    int j     = kk & 15;
    int lane  = khalf * 16 + (n & 15);
    int tile  = n >> 4;
    __bf16* dst = (__bf16*)ws;
    dst[(size_t)w * (size_t)(K96 * GATE) +
        (size_t)(((c * 25 + tile) * 32 + lane) * 16 + j)] = (__bf16)v;
  } else if (idx < 3 * K96 * GATE + 3 * 4 * GATE) {
    int r = idx - 3 * K96 * GATE;
    int w = r / (4 * GATE);
    int q = (r % (4 * GATE)) / GATE;
    int n = r % GATE;
    const float* W = (w == 0) ? whh1 : (w == 1) ? wih2 : whh2;
    float* dst = (float*)(ws + LDS_WTAIL);
    dst[(w * 4 + q) * GATE + n] = W[n * HDIM + (K96 + q)];
  }
}

// ---------------------------------------------------------------------------
// Main persistent fused LSTM kernel.
// ---------------------------------------------------------------------------
extern "C" __global__ void __launch_bounds__(256)
lstm_main(const float* __restrict__ x,
          const float* __restrict__ wih1,
          const float* __restrict__ bih1, const float* __restrict__ bhh1,
          const float* __restrict__ bih2, const float* __restrict__ bhh2,
          const float* __restrict__ fc1w, const float* __restrict__ fc1b,
          const float* __restrict__ fc2w, const float* __restrict__ fc2b,
          const char*  __restrict__ ws,   float* __restrict__ out)
{
  extern __shared__ char smem[];
  const int tid  = threadIdx.x;
  const int lane = tid & 31;
  const int wid  = tid >> 5;                 // wave id (wave32), 8 waves
  const int b0   = blockIdx.x * 16;          // batch row base for this WG

  // ---- Stage weights ws -> LDS ----
#if HAVE_TDM
  // Fragment region (230400 B) as one 1-D TDM transfer: 28800 x 8B elements.
  if (wid == 0) {
    const unsigned long long ga = (unsigned long long)(uintptr_t)ws;
    const uint32_t nelem = WFRAG_BYTES / 8u;           // 28800
    v4u g0 = { 1u,                                     // count=1 (valid D#)
               0u,                                     // lds_addr = 0
               (uint32_t)ga,                           // global_addr[31:0]
               (uint32_t)((ga >> 32) & 0x1FFFFFFu) | (2u << 30) }; // [56:32] | type=2
    v8i g1 = { (int)(3u << 16),                        // data_size = 8B
               (int)((nelem & 0xFFFFu) << 16),         // tensor_dim0[15:0]
               (int)(((nelem >> 16) & 0xFFFFu) | (1u << 16)), // dim0[31:16], tensor_dim1=1
               (int)((nelem & 0xFFFFu) << 16),         // tile_dim0
               1,                                      // tile_dim1 = 1
               (int)nelem,                             // tensor_dim0_stride[31:0]
               0, 0 };
    v4i gz = {0, 0, 0, 0};
#if defined(__clang_major__) && (__clang_major__ >= 23)
    v8i gz8 = {0, 0, 0, 0, 0, 0, 0, 0};
    __builtin_amdgcn_tensor_load_to_lds(g0, g1, gz, gz, gz8, 0);
#else
    __builtin_amdgcn_tensor_load_to_lds(g0, g1, gz, gz, 0);
#endif
  }
  // Tail region (19200 B) via per-lane async copies (ASYNCcnt path).
  {
    const unsigned long long base = (unsigned long long)(uintptr_t)ws;
    #pragma unroll
    for (uint32_t i = 0; i < 5; ++i) {
      uint32_t idx = (uint32_t)tid + 256u * i;
      if (idx < (WS_WBYTES - WFRAG_BYTES) / 16u) {
        uint32_t off = WFRAG_BYTES + idx * 16u;
        asm volatile("global_load_async_to_lds_b128 %0, %1, %2"
                     :: "v"(off), "v"(off), "s"(base));
      }
    }
  }
#else
  // Fallback: whole region via async copies.
  {
    const unsigned long long base = (unsigned long long)(uintptr_t)ws;
    #pragma unroll 4
    for (uint32_t i = 0; i < 61; ++i) {
      uint32_t off = (uint32_t)(tid + 256u * i) * 16u;
      if (off < WS_WBYTES) {
        asm volatile("global_load_async_to_lds_b128 %0, %1, %2"
                     :: "v"(off), "v"(off), "s"(base));
      }
    }
  }
#endif

  // wih1 (400x8 f32) -> LDS
  for (int i = tid; i < (GATE * 8) / 4; i += 256) {
    v4f v = ((const v4f*)wih1)[i];
    *(v4f*)(smem + LDS_WIH1 + (size_t)i * 16) = v;
  }
  // combined biases
  for (int i = tid; i < GATE; i += 256) {
    *(float*)(smem + LDS_B1 + (size_t)i * 4) = bih1[i] + bhh1[i];
    *(float*)(smem + LDS_B2 + (size_t)i * 4) = bih2[i] + bhh2[i];
  }
  // zero h/c state (f32) and bf16 A-fragment h buffers
  for (int i = tid; i < 16 * HDIM; i += 256) {
    *(float*)(smem + LDS_H1F + (size_t)i * 4) = 0.f;
    *(float*)(smem + LDS_C1  + (size_t)i * 4) = 0.f;
    *(float*)(smem + LDS_H2F + (size_t)i * 4) = 0.f;
    *(float*)(smem + LDS_C2  + (size_t)i * 4) = 0.f;
  }
  for (int i = tid; i < 768; i += 256) {
    *(uint32_t*)(smem + LDS_H1B + (size_t)i * 4) = 0u;
    *(uint32_t*)(smem + LDS_H2B + (size_t)i * 4) = 0u;
  }

  asm volatile("s_wait_asynccnt 0" ::: "memory");
#if HAVE_TDM
#if __has_builtin(__builtin_amdgcn_s_wait_tensorcnt)
  __builtin_amdgcn_s_wait_tensorcnt(0);
#else
  asm volatile("s_wait_tensorcnt 0" ::: "memory");
#endif
#endif
  __syncthreads();

  const int mhi = (lane >> 4) << 3;      // 0 or 8 (C/D M offset)
  const int nlo = lane & 15;             // C/D column within tile

  for (int t = 0; t < TLEN; ++t) {
    // ---- Phase 1: stage x[b0:b0+16, t, 0:8] + prefetch ahead ----
    if (tid < 32) {
      int m = tid >> 1, hh = tid & 1;
      v4f v = *(const v4f*)(x + ((size_t)(b0 + m) * TLEN + t) * 8 + hh * 4);
      *(v4f*)(smem + LDS_XROW + (size_t)(m * 8 + hh * 4) * 4) = v;
      if (t + 16 < TLEN)
        __builtin_prefetch(x + ((size_t)(b0 + m) * TLEN + (t + 16)) * 8, 0, 1);
    }
    __syncthreads();

    // ---- Phase 2: layer-1 gates  g = b1 + x·Wih1ᵀ + h1·Whh1ᵀ ----
    {
      v16bf a[3];
      #pragma unroll
      for (int c = 0; c < 3; ++c)
        a[c] = *(const v16bf*)(smem + LDS_H1B + (size_t)(c * 1024 + lane * 32));
      for (int tl = wid; tl < 25; tl += 8) {
        v8f acc = {0.f, 0.f, 0.f, 0.f, 0.f, 0.f, 0.f, 0.f};
        #pragma unroll
        for (int c = 0; c < 3; ++c) {
          v16bf bb = *(const v16bf*)(smem + LDS_WFRAG +
                        (size_t)(((c * 25 + tl) * 32 + lane)) * 32u);
          acc = __builtin_amdgcn_wmma_f32_16x16x32_bf16(
                  false, a[c], false, bb, (short)0, acc, false, false);
        }
        int n = tl * 16 + nlo;
        float bias = *(const float*)(smem + LDS_B1 + (size_t)n * 4);
        v4f w0 = *(const v4f*)(smem + LDS_WIH1 + (size_t)n * 32);
        v4f w1 = *(const v4f*)(smem + LDS_WIH1 + (size_t)n * 32 + 16);
        v4f tw = { *(const float*)(smem + LDS_WTAIL + (size_t)((0 * 4 + 0) * GATE + n) * 4),
                   *(const float*)(smem + LDS_WTAIL + (size_t)((0 * 4 + 1) * GATE + n) * 4),
                   *(const float*)(smem + LDS_WTAIL + (size_t)((0 * 4 + 2) * GATE + n) * 4),
                   *(const float*)(smem + LDS_WTAIL + (size_t)((0 * 4 + 3) * GATE + n) * 4) };
        #pragma unroll
        for (int r = 0; r < 8; ++r) {
          int m = r + mhi;
          v4f x0 = *(const v4f*)(smem + LDS_XROW + (size_t)m * 32);
          v4f x1 = *(const v4f*)(smem + LDS_XROW + (size_t)m * 32 + 16);
          v4f th = *(const v4f*)(smem + LDS_H1F + (size_t)(m * HDIM + 96) * 4);
          float g = acc[r] + bias
                  + x0[0]*w0[0] + x0[1]*w0[1] + x0[2]*w0[2] + x0[3]*w0[3]
                  + x1[0]*w1[0] + x1[1]*w1[1] + x1[2]*w1[2] + x1[3]*w1[3]
                  + th[0]*tw[0] + th[1]*tw[1] + th[2]*tw[2] + th[3]*tw[3];
          *(float*)(smem + LDS_G + (size_t)(m * GATE + n) * 4) = g;
        }
      }
    }
    __syncthreads();

    // ---- Phase 3: layer-1 elementwise ----
    {
      int m = tid >> 4, s = tid & 15;
      for (int j = s; j < HDIM; j += 16) {
        float gi = *(const float*)(smem + LDS_G + (size_t)(m * GATE +       j) * 4);
        float gf = *(const float*)(smem + LDS_G + (size_t)(m * GATE + 100 + j) * 4);
        float gc = *(const float*)(smem + LDS_G + (size_t)(m * GATE + 200 + j) * 4);
        float go = *(const float*)(smem + LDS_G + (size_t)(m * GATE + 300 + j) * 4);
        float cold = *(const float*)(smem + LDS_C1 + (size_t)(m * HDIM + j) * 4);
        float cnew = sigm(gf) * cold + sigm(gi) * tanh_fast(gc);
        float hnew = sigm(go) * tanh_fast(cnew);
        *(float*)(smem + LDS_C1  + (size_t)(m * HDIM + j) * 4) = cnew;
        *(float*)(smem + LDS_H1F + (size_t)(m * HDIM + j) * 4) = hnew;
        if (j < K96) {     // A-fragment-order bf16 store (k = j)
          int kk = j & 31, c = j >> 5;
          int lane_w = m + (kk & 16);
          int elem   = kk & 15;
          *(__bf16*)(smem + LDS_H1B + (size_t)(c * 1024 + lane_w * 32 + elem * 2)) = (__bf16)hnew;
        }
      }
    }
    __syncthreads();

    // ---- Phase 4: layer-2 gates  g = b2 + h1ₜ·Wih2ᵀ + h2·Whh2ᵀ ----
    {
      v16bf a1[3], a2[3];
      #pragma unroll
      for (int c = 0; c < 3; ++c) {
        a1[c] = *(const v16bf*)(smem + LDS_H1B + (size_t)(c * 1024 + lane * 32));
        a2[c] = *(const v16bf*)(smem + LDS_H2B + (size_t)(c * 1024 + lane * 32));
      }
      for (int tl = wid; tl < 25; tl += 8) {
        v8f acc = {0.f, 0.f, 0.f, 0.f, 0.f, 0.f, 0.f, 0.f};
        #pragma unroll
        for (int c = 0; c < 3; ++c) {
          v16bf bb = *(const v16bf*)(smem + LDS_WFRAG + 1u * 76800u +
                        (size_t)(((c * 25 + tl) * 32 + lane)) * 32u);
          acc = __builtin_amdgcn_wmma_f32_16x16x32_bf16(
                  false, a1[c], false, bb, (short)0, acc, false, false);
        }
        #pragma unroll
        for (int c = 0; c < 3; ++c) {
          v16bf bb = *(const v16bf*)(smem + LDS_WFRAG + 2u * 76800u +
                        (size_t)(((c * 25 + tl) * 32 + lane)) * 32u);
          acc = __builtin_amdgcn_wmma_f32_16x16x32_bf16(
                  false, a2[c], false, bb, (short)0, acc, false, false);
        }
        int n = tl * 16 + nlo;
        float bias = *(const float*)(smem + LDS_B2 + (size_t)n * 4);
        v4f tw1 = { *(const float*)(smem + LDS_WTAIL + (size_t)((1 * 4 + 0) * GATE + n) * 4),
                    *(const float*)(smem + LDS_WTAIL + (size_t)((1 * 4 + 1) * GATE + n) * 4),
                    *(const float*)(smem + LDS_WTAIL + (size_t)((1 * 4 + 2) * GATE + n) * 4),
                    *(const float*)(smem + LDS_WTAIL + (size_t)((1 * 4 + 3) * GATE + n) * 4) };
        v4f tw2 = { *(const float*)(smem + LDS_WTAIL + (size_t)((2 * 4 + 0) * GATE + n) * 4),
                    *(const float*)(smem + LDS_WTAIL + (size_t)((2 * 4 + 1) * GATE + n) * 4),
                    *(const float*)(smem + LDS_WTAIL + (size_t)((2 * 4 + 2) * GATE + n) * 4),
                    *(const float*)(smem + LDS_WTAIL + (size_t)((2 * 4 + 3) * GATE + n) * 4) };
        #pragma unroll
        for (int r = 0; r < 8; ++r) {
          int m = r + mhi;
          v4f t1 = *(const v4f*)(smem + LDS_H1F + (size_t)(m * HDIM + 96) * 4);
          v4f t2 = *(const v4f*)(smem + LDS_H2F + (size_t)(m * HDIM + 96) * 4);
          float g = acc[r] + bias
                  + t1[0]*tw1[0] + t1[1]*tw1[1] + t1[2]*tw1[2] + t1[3]*tw1[3]
                  + t2[0]*tw2[0] + t2[1]*tw2[1] + t2[2]*tw2[2] + t2[3]*tw2[3];
          *(float*)(smem + LDS_G + (size_t)(m * GATE + n) * 4) = g;
        }
      }
    }
    __syncthreads();

    // ---- Phase 5: layer-2 elementwise ----
    {
      int m = tid >> 4, s = tid & 15;
      for (int j = s; j < HDIM; j += 16) {
        float gi = *(const float*)(smem + LDS_G + (size_t)(m * GATE +       j) * 4);
        float gf = *(const float*)(smem + LDS_G + (size_t)(m * GATE + 100 + j) * 4);
        float gc = *(const float*)(smem + LDS_G + (size_t)(m * GATE + 200 + j) * 4);
        float go = *(const float*)(smem + LDS_G + (size_t)(m * GATE + 300 + j) * 4);
        float cold = *(const float*)(smem + LDS_C2 + (size_t)(m * HDIM + j) * 4);
        float cnew = sigm(gf) * cold + sigm(gi) * tanh_fast(gc);
        float hnew = sigm(go) * tanh_fast(cnew);
        *(float*)(smem + LDS_C2  + (size_t)(m * HDIM + j) * 4) = cnew;
        *(float*)(smem + LDS_H2F + (size_t)(m * HDIM + j) * 4) = hnew;
        if (j < K96) {
          int kk = j & 31, c = j >> 5;
          int lane_w = m + (kk & 16);
          int elem   = kk & 15;
          *(__bf16*)(smem + LDS_H2B + (size_t)(c * 1024 + lane_w * 32 + elem * 2)) = (__bf16)hnew;
        }
      }
    }
    __syncthreads();
  }

  // ---- FC head on h2[:, T-1, :]  (tiny; VALU from LDS) ----
  for (int e = tid; e < 16 * 25; e += 256) {
    int m = e / 25, p = e % 25;
    float acc = fc1b[p];
    for (int k = 0; k < HDIM; ++k)
      acc += *(const float*)(smem + LDS_H2F + (size_t)(m * HDIM + k) * 4) * fc1w[p * HDIM + k];
    *(float*)(smem + LDS_G + (size_t)(m * 25 + p) * 4) = acc;   // reuse g buffer
  }
  __syncthreads();
  if (tid < 16) {
    float acc = fc2b[0];
    for (int p = 0; p < 25; ++p)
      acc += *(const float*)(smem + LDS_G + (size_t)(tid * 25 + p) * 4) * fc2w[p];
    out[b0 + tid] = acc;
  }
}

// ---------------------------------------------------------------------------
extern "C" void kernel_launch(void* const* d_in, const int* in_sizes, int n_in,
                              void* d_out, int out_size, void* d_ws, size_t ws_size,
                              hipStream_t stream) {
  const float* x    = (const float*)d_in[0];
  const float* wih1 = (const float*)d_in[1];
  const float* whh1 = (const float*)d_in[2];
  const float* bih1 = (const float*)d_in[3];
  const float* bhh1 = (const float*)d_in[4];
  const float* wih2 = (const float*)d_in[5];
  const float* whh2 = (const float*)d_in[6];
  const float* bih2 = (const float*)d_in[7];
  const float* bhh2 = (const float*)d_in[8];
  const float* fc1w = (const float*)d_in[9];
  const float* fc1b = (const float*)d_in[10];
  const float* fc2w = (const float*)d_in[11];
  const float* fc2b = (const float*)d_in[12];

  char* ws = (char*)d_ws;
  const int prepElems = 3 * K96 * GATE + 3 * 4 * GATE;   // 120000
  lstm_prep<<<(prepElems + 255) / 256, 256, 0, stream>>>(whh1, wih2, whh2, ws);
  lstm_main<<<16, 256, LDS_TOTAL, stream>>>(x, wih1, bih1, bhh1, bih2, bhh2,
                                            fc1w, fc1b, fc2w, fc2b,
                                            (const char*)ws, (float*)d_out);
}
